// synandred_81982335746417
// MI455X (gfx1250) — matrix operand order repaired and checked by reading, not compiled, not run on hardware
//
#include <hip/hip_runtime.h>
#include <hip/hip_bf16.h>

// ---------------------------------------------------------------------------
// Problem constants (from reference): B=8192 rows, N=2048 confounders, D=1024
// ---------------------------------------------------------------------------
constexpr int Bq = 8192;
constexpr int Nc = 2048;
constexpr int Dd = 1024;

typedef __attribute__((ext_vector_type(16))) __bf16 bf16x16;
typedef __attribute__((ext_vector_type(8)))  __bf16 bf16x8;
typedef __attribute__((ext_vector_type(4)))  __bf16 bf16x4;
typedef __attribute__((ext_vector_type(8)))  float  f32x8;

typedef unsigned int u32;
typedef u32 __attribute__((ext_vector_type(4))) u32x4;
typedef int __attribute__((ext_vector_type(4))) i32x4;
typedef int __attribute__((ext_vector_type(8))) i32x8;

#define HAVE_TDM __has_builtin(__builtin_amdgcn_tensor_load_to_lds)

// Wait for all of this wave's async (global->LDS) loads to complete.
static __device__ __forceinline__ void wait_async0() {
#if __has_builtin(__builtin_amdgcn_s_wait_asynccnt)
    __builtin_amdgcn_s_wait_asynccnt(0);
#else
    asm volatile("s_wait_asynccnt 0" ::: "memory");
#endif
}

// Wait for all of this wave's TDM transfers to complete.
static __device__ __forceinline__ void wait_tensor0() {
#if __has_builtin(__builtin_amdgcn_s_wait_tensorcnt)
    __builtin_amdgcn_s_wait_tensorcnt(0);
#else
    asm volatile("s_wait_tensorcnt 0" ::: "memory");
#endif
}

// One 16-byte async copy global -> LDS (GLOBAL_LOAD_ASYNC_TO_LDS_B128,
// ASYNCcnt-tracked; no VGPR staging). Fallback path when TDM is unavailable.
static __device__ __forceinline__ void async_copy_b128(unsigned int lds_off,
                                                       const void* gptr) {
    unsigned long long ga = (unsigned long long)(uintptr_t)gptr;
    asm volatile("global_load_async_to_lds_b128 %0, %1, off"
                 :: "v"(lds_off), "v"(ga) : "memory");
}

#if HAVE_TDM
// Tensor Data Mover: DMA one 2D bf16 tile [tile_d1 rows x tile_d0 elems]
// (row stride = stride0 elems in memory) into LDS at lds_addr, inserting
// 16 B of padding after every 64 B row (pad_interval=3 -> every 16 DWORDs,
// pad_amount=3 -> 4 DWORDs), i.e. an 80 B LDS row pitch == LDSW bf16.
// D# bit packing per CDNA5 ISA 8.3/8.4; TENSORcnt-tracked.
// This toolchain's builtin takes 6 args:
//   (u32x4 group0, i32x8 group1, i32x4 group2, i32x4 group3, i32x8, i32 cpol)
static __device__ __forceinline__ void tdm_load_tile_2d(
    u32 lds_addr, const void* gptr, u32 tensor_d0, u32 tensor_d1,
    u32 tile_d0, u32 tile_d1, u32 stride0) {
    unsigned long long ga = (unsigned long long)(uintptr_t)gptr;
    u32x4 g0;
    g0.x = 1u;                                  // count=1 (valid), user mode
    g0.y = lds_addr;                            // lds_addr[31:0]
    g0.z = (u32)ga;                             // global_addr[31:0]
    g0.w = (u32)(ga >> 32) | (2u << 30);        // global_addr[56:32] | type=2
    i32x8 g1;
    g1[0] = (int)((1u << 16)        // data_size = 1 -> 2-byte elements
                | (1u << 20)        // pad_enable
                | (3u << 22)        // pad_interval: every 16 DWORDs (64 B)
                | (3u << 25));      // pad_amount: 4 DWORDs (16 B)
    g1[1] = (int)((tensor_d0 & 0xffffu) << 16);                    // dim0[15:0]
    g1[2] = (int)((tensor_d0 >> 16) | ((tensor_d1 & 0xffffu) << 16));
    g1[3] = (int)((tensor_d1 >> 16) | (tile_d0 << 16));
    g1[4] = (int)tile_d1;                       // tile_dim1; tile_dim2 = 0
    g1[5] = (int)stride0;                       // tensor_dim0_stride[31:0]
    g1[6] = 0;                                  // stride0[47:32], dim1_stride lo
    g1[7] = 0;
    i32x4 gz4 = {0, 0, 0, 0};
    i32x8 gz8 = {0, 0, 0, 0, 0, 0, 0, 0};
    __builtin_amdgcn_tensor_load_to_lds(g0, g1, gz4, gz4, gz8, 0);
}
#endif

// ---------------------------------------------------------------------------
// f32 -> bf16 bulk convert (vectorized b128 load / b64 store)
// ---------------------------------------------------------------------------
__global__ void __launch_bounds__(256)
cvt_f32_to_bf16(const float4* __restrict__ src, bf16x4* __restrict__ dst, int n4) {
    int i = blockIdx.x * 256 + threadIdx.x;
    if (i < n4) {
        float4 v = src[i];
        bf16x4 o = { (__bf16)v.x, (__bf16)v.y, (__bf16)v.z, (__bf16)v.w };
        dst[i] = o;
    }
}

// ---------------------------------------------------------------------------
// Vt[d][n] = Cd[n][d] * prior[n]   (transpose through LDS so the attn@V GEMM
// is NT like everything else; coalesced both ways)
// ---------------------------------------------------------------------------
__global__ void __launch_bounds__(256)
transpose_scale_bf16(const float* __restrict__ Cd, const float* __restrict__ prior,
                     __bf16* __restrict__ Vt, int N, int D) {
    __shared__ float tile[32][33];
    int n0 = blockIdx.y * 32, d0 = blockIdx.x * 32;
    int tx = threadIdx.x & 31, ty = threadIdx.x >> 5;   // ty = 0..7
#pragma unroll
    for (int r = 0; r < 32; r += 8) {
        int n = n0 + ty + r;
        tile[ty + r][tx] = Cd[(size_t)n * D + d0 + tx] * prior[n];
    }
    __syncthreads();
#pragma unroll
    for (int r = 0; r < 32; r += 8) {
        int d = d0 + ty + r;
        Vt[(size_t)d * N + n0 + tx] = (__bf16)tile[tx][ty + r];
    }
}

// ---------------------------------------------------------------------------
// Tiled NT GEMM:  C[M,N] = scale * (A[M,K] @ B[N,K]^T) (+ bias[N])
// A,B bf16 row-major (K contiguous) -> WMMA fragments in contiguous k order (k = 16*half + s).
// Block tile 128(M) x 256(N) x 32(K), 256 threads = 8 waves in a 2(M) x 4(N)
// grid; each wave owns a 64x64 tile = 4x4 C fragments of 16x16
// (16 v_wmma per K-step per wave, 16 ds_load_b128 fragment fetches).
// Tiles staged by the Tensor Data Mover (one tensor_load_to_lds per tile,
// TENSORcnt) with TDM padding producing the 80 B LDS row pitch; double
// buffered. Fallback: per-lane GLOBAL_LOAD_ASYNC_TO_LDS_B128 (ASYNCcnt).
// ---------------------------------------------------------------------------
#define BM 128
#define BN 256
#define BK 32
#define LDSW 40   // 40 bf16 = 80 bytes (multiple of 16B; 20-bank step, conflict free)

template <bool OUT_BF16, bool ADD_BIAS>
__global__ void __launch_bounds__(256)
gemm_nt_bf16(const __bf16* __restrict__ A, const __bf16* __restrict__ B,
             void* __restrict__ Cout, const float* __restrict__ bias,
             int M, int N, int K, float scale) {
    __shared__ __align__(16) __bf16 sA[2][BM][LDSW];
    __shared__ __align__(16) __bf16 sB[2][BN][LDSW];

    const int tid  = threadIdx.x;
    const int lane = tid & 31;
    const int wave = tid >> 5;
    const int wm   = wave >> 2;         // 0..1 -> M offset wm*64
    const int wn   = wave & 3;          // 0..3 -> N offset wn*64
    const int m_blk = blockIdx.y * BM;
    const int n_blk = blockIdx.x * BN;
    const int half = lane >> 4;         // 0: lanes 0-15, 1: lanes 16-31
    const int l15  = lane & 15;

    f32x8 acc[4][4];
    const f32x8 zero = {0.f,0.f,0.f,0.f,0.f,0.f,0.f,0.f};
#pragma unroll
    for (int i = 0; i < 4; ++i)
#pragma unroll
        for (int j = 0; j < 4; ++j) acc[i][j] = zero;

    // Stage one 128x32 A tile and one 256x32 B tile into LDS buffer buf.
    auto stage_tile = [&](int buf, int k0) {
#if HAVE_TDM
        if (wave == 0) {   // one TDM descriptor pair per tile for the block
            tdm_load_tile_2d((u32)(uintptr_t)&sA[buf][0][0],
                             A + (size_t)m_blk * K + k0,
                             (u32)K, (u32)M, (u32)BK, (u32)BM, (u32)K);
            tdm_load_tile_2d((u32)(uintptr_t)&sB[buf][0][0],
                             B + (size_t)n_blk * K + k0,
                             (u32)K, (u32)N, (u32)BK, (u32)BN, (u32)K);
        }
#else
#pragma unroll
        for (int it = 0; it < 2; ++it) {            // A: 512 x 16B segments
            int seg = tid + it * 256;
            int row = seg >> 2;
            int col = (seg & 3) << 3;
            async_copy_b128((unsigned int)(uintptr_t)&sA[buf][row][col],
                            A + (size_t)(m_blk + row) * K + k0 + col);
        }
#pragma unroll
        for (int it = 0; it < 4; ++it) {            // B: 1024 x 16B segments
            int seg = tid + it * 256;
            int row = seg >> 2;
            int col = (seg & 3) << 3;
            async_copy_b128((unsigned int)(uintptr_t)&sB[buf][row][col],
                            B + (size_t)(n_blk + row) * K + k0 + col);
        }
#endif
    };
    auto stage_wait = [&]() {
#if HAVE_TDM
        if (wave == 0) wait_tensor0();   // TENSORcnt is per-wave; issuer waits
#else
        wait_async0();
#endif
    };

    stage_tile(0, 0);
    stage_wait();
    __syncthreads();

    const int nk = K / BK;
    for (int kt = 0; kt < nk; ++kt) {
        const int buf = kt & 1;
        if (kt + 1 < nk) stage_tile(buf ^ 1, (kt + 1) * BK);

        // B fragments: 32x16 bf16. Lanes 0-15 hold K 0..15 of col N=lane,
        // lanes 16-31 hold K 16..31 of col N=lane-16. Contiguous per lane.
        bf16x16 bfrag[4];
#pragma unroll
        for (int j = 0; j < 4; ++j) {
            int col = wn * 64 + j * 16 + l15;
            union { bf16x16 v; bf16x8 h[2]; } u;
            u.h[0] = *(const bf16x8*)__builtin_assume_aligned(&sB[buf][col][half * 16], 16);
            u.h[1] = *(const bf16x8*)__builtin_assume_aligned(&sB[buf][col][half * 16 + 8], 16);
            bfrag[j] = u.v;
        }
        // A fragments: 16x32 bf16. Lanes 0-15 hold K 0..15,
        // lanes 16-31 hold K 16..31 for row M=lane&15. Contiguous per lane.
#pragma unroll
        for (int i = 0; i < 4; ++i) {
            int row = wm * 64 + i * 16 + l15;
            union { bf16x16 v; bf16x8 h[2]; } u;
            u.h[0] = *(const bf16x8*)__builtin_assume_aligned(&sA[buf][row][half * 16], 16);
            u.h[1] = *(const bf16x8*)__builtin_assume_aligned(&sA[buf][row][half * 16 + 8], 16);
            bf16x16 afrag = u.v;
#pragma unroll
            for (int j = 0; j < 4; ++j)
                acc[i][j] = __builtin_amdgcn_wmma_f32_16x16x32_bf16(
                    false, afrag, false, bfrag[j], (short)0, acc[i][j], false, false);
        }
        stage_wait();        // next tile resident in LDS
        __syncthreads();     // all waves done reading buf / writes to buf^1 visible
    }

    // Epilogue. C frag layout: VGPR v -> row (half*8 + v), col = lane&15.
#pragma unroll
    for (int i = 0; i < 4; ++i) {
#pragma unroll
        for (int j = 0; j < 4; ++j) {
            int col = n_blk + wn * 64 + j * 16 + l15;
            float bv = ADD_BIAS ? bias[col] : 0.0f;
#pragma unroll
            for (int v = 0; v < 8; ++v) {
                int row = m_blk + wm * 64 + i * 16 + half * 8 + v;
                float val = acc[i][j][v] * scale + bv;
                if (OUT_BF16)
                    ((__bf16*)Cout)[(size_t)row * N + col] = (__bf16)val;
                else
                    ((float*)Cout)[(size_t)row * N + col] = val;
            }
        }
    }
}

// ---------------------------------------------------------------------------
// Row softmax: one wave per row of S[rows, 2048]; all 64 elems/lane live in
// registers between the max and exp/sum passes. Output bf16 (feeds P@V GEMM).
// ---------------------------------------------------------------------------
__global__ void __launch_bounds__(256)
softmax_rows(const float* __restrict__ S, __bf16* __restrict__ P, int cols) {
    const int wave = threadIdx.x >> 5, lane = threadIdx.x & 31;
    const int row = blockIdx.x * 8 + wave;
    const float4* s = (const float4*)(S + (size_t)row * cols);

    float4 v[16];
    float mx = -3.4e38f;
#pragma unroll
    for (int i = 0; i < 16; ++i) {
        v[i] = s[lane + i * 32];
        mx = fmaxf(mx, fmaxf(fmaxf(v[i].x, v[i].y), fmaxf(v[i].z, v[i].w)));
    }
#pragma unroll
    for (int o = 16; o > 0; o >>= 1) mx = fmaxf(mx, __shfl_xor(mx, o, 32));

    float sum = 0.f;
#pragma unroll
    for (int i = 0; i < 16; ++i) {
        v[i].x = __expf(v[i].x - mx); v[i].y = __expf(v[i].y - mx);
        v[i].z = __expf(v[i].z - mx); v[i].w = __expf(v[i].w - mx);
        sum += v[i].x + v[i].y + v[i].z + v[i].w;
    }
#pragma unroll
    for (int o = 16; o > 0; o >>= 1) sum += __shfl_xor(sum, o, 32);

    const float inv = 1.0f / sum;
    bf16x4* p = (bf16x4*)(P + (size_t)row * cols);
#pragma unroll
    for (int i = 0; i < 16; ++i) {
        bf16x4 o = { (__bf16)(v[i].x * inv), (__bf16)(v[i].y * inv),
                     (__bf16)(v[i].z * inv), (__bf16)(v[i].w * inv) };
        p[lane + i * 32] = o;
    }
}

// ---------------------------------------------------------------------------
// Final: y = (1-sigmoid(bal))*jf_aug + sigmoid(bal)*G, then LayerNorm over D.
// One wave per row; D=1024 -> 8 float4 per lane.
// ---------------------------------------------------------------------------
__global__ void __launch_bounds__(256)
combine_layernorm(const float* __restrict__ G, const float* __restrict__ jf_aug,
                  const float* __restrict__ balance, const float* __restrict__ gamma,
                  const float* __restrict__ beta, float* __restrict__ out, int D) {
    const int wave = threadIdx.x >> 5, lane = threadIdx.x & 31;
    const int row = blockIdx.x * 8 + wave;
    const float4* g = (const float4*)(G + (size_t)row * D);
    const float4* a = (const float4*)(jf_aug + (size_t)row * D);
    const float4* bl = (const float4*)balance;

    float4 y[8];
    float s = 0.f, ss = 0.f;
#pragma unroll
    for (int i = 0; i < 8; ++i) {
        int idx = lane + i * 32;
        float4 gv = g[idx], av = a[idx], bv = bl[idx];
        float wx = 1.f / (1.f + __expf(-bv.x));
        float wy = 1.f / (1.f + __expf(-bv.y));
        float wz = 1.f / (1.f + __expf(-bv.z));
        float ww = 1.f / (1.f + __expf(-bv.w));
        y[i].x = (1.f - wx) * av.x + wx * gv.x;
        y[i].y = (1.f - wy) * av.y + wy * gv.y;
        y[i].z = (1.f - wz) * av.z + wz * gv.z;
        y[i].w = (1.f - ww) * av.w + ww * gv.w;
        s  += y[i].x + y[i].y + y[i].z + y[i].w;
        ss += y[i].x * y[i].x + y[i].y * y[i].y + y[i].z * y[i].z + y[i].w * y[i].w;
    }
#pragma unroll
    for (int o = 16; o > 0; o >>= 1) { s += __shfl_xor(s, o, 32); ss += __shfl_xor(ss, o, 32); }

    const float mean = s / (float)D;
    const float var  = ss / (float)D - mean * mean;
    const float rstd = rsqrtf(var + 1e-5f);

    float4* orow = (float4*)(out + (size_t)row * D);
    const float4* gm = (const float4*)gamma;
    const float4* bt = (const float4*)beta;
#pragma unroll
    for (int i = 0; i < 8; ++i) {
        int idx = lane + i * 32;
        float4 gmv = gm[idx], btv = bt[idx], o;
        o.x = (y[i].x - mean) * rstd * gmv.x + btv.x;
        o.y = (y[i].y - mean) * rstd * gmv.y + btv.y;
        o.z = (y[i].z - mean) * rstd * gmv.z + btv.z;
        o.w = (y[i].w - mean) * rstd * gmv.w + btv.w;
        orow[idx] = o;
    }
}

// ---------------------------------------------------------------------------
// Launch pipeline
// ---------------------------------------------------------------------------
extern "C" void kernel_launch(void* const* d_in, const int* in_sizes, int n_in,
                              void* d_out, int out_size, void* d_ws, size_t ws_size,
                              hipStream_t stream) {
    (void)in_sizes; (void)n_in; (void)out_size; (void)ws_size;
    const float* jf      = (const float*)d_in[0];   // (8192,1,1024)
    const float* Cd      = (const float*)d_in[1];   // (2048,1024)
    const float* prior   = (const float*)d_in[2];   // (2048,1)
    const float* Wq      = (const float*)d_in[3];   // (1024,1024)
    const float* Wk      = (const float*)d_in[4];
    const float* augW    = (const float*)d_in[5];
    const float* augb    = (const float*)d_in[6];   // (1024,)
    const float* balance = (const float*)d_in[7];
    const float* gamma   = (const float*)d_in[8];
    const float* beta    = (const float*)d_in[9];
    float* out = (float*)d_out;

    // Workspace carve-up (256B aligned slabs). ~185 MB total.
    char* ws = (char*)d_ws;
    size_t off = 0;
    auto carve = [&](size_t bytes) -> char* {
        char* p = ws + off;
        off += (bytes + 255) & ~(size_t)255;
        return p;
    };
    __bf16* jf_b   = (__bf16*)carve((size_t)Bq * Dd * 2);   // 16 MB
    __bf16* Cd_b   = (__bf16*)carve((size_t)Nc * Dd * 2);   //  4 MB
    __bf16* Wq_b   = (__bf16*)carve((size_t)Dd * Dd * 2);   //  2 MB
    __bf16* Wk_b   = (__bf16*)carve((size_t)Dd * Dd * 2);   //  2 MB
    __bf16* augW_b = (__bf16*)carve((size_t)Dd * Dd * 2);   //  2 MB
    __bf16* Vt_b   = (__bf16*)carve((size_t)Dd * Nc * 2);   //  4 MB  (V^T)
    __bf16* K_b    = (__bf16*)carve((size_t)Nc * Dd * 2);   //  4 MB
    __bf16* Q_b    = (__bf16*)carve((size_t)Bq * Dd * 2);   // 16 MB
    float*  jf_aug = (float*) carve((size_t)Bq * Dd * 4);   // 32 MB
    float*  S      = (float*) carve((size_t)Bq * Nc * 4);   // 64 MB
    __bf16* attn   = (__bf16*)carve((size_t)Bq * Nc * 2);   // 32 MB
    float*  G      = S;   // alias: S is dead after softmax; G is 32 MB < 64 MB

    // 1) f32 -> bf16 converts
    auto cvt = [&](const float* src, __bf16* dst, size_t n) {
        int n4 = (int)(n / 4);
        cvt_f32_to_bf16<<<(n4 + 255) / 256, 256, 0, stream>>>(
            (const float4*)src, (bf16x4*)dst, n4);
    };
    cvt(jf,   jf_b,   (size_t)Bq * Dd);
    cvt(Cd,   Cd_b,   (size_t)Nc * Dd);
    cvt(Wq,   Wq_b,   (size_t)Dd * Dd);
    cvt(Wk,   Wk_b,   (size_t)Dd * Dd);
    cvt(augW, augW_b, (size_t)Dd * Dd);

    // 2) Vt[d][n] = Cd[n][d] * prior[n]
    transpose_scale_bf16<<<dim3(Dd / 32, Nc / 32), 256, 0, stream>>>(Cd, prior, Vt_b, Nc, Dd);

    const float rsqrtD = 0.03125f;  // 1/sqrt(1024)

    // 3) K = Cd @ Wk^T           (2048 x 1024, K=1024) -> bf16
    gemm_nt_bf16<true, false><<<dim3(Dd / BN, Nc / BM), 256, 0, stream>>>(
        Cd_b, Wk_b, K_b, nullptr, Nc, Dd, Dd, 1.0f);

    // 4) Q = jf @ Wq^T           (8192 x 1024, K=1024) -> bf16
    gemm_nt_bf16<true, false><<<dim3(Dd / BN, Bq / BM), 256, 0, stream>>>(
        jf_b, Wq_b, Q_b, nullptr, Bq, Dd, Dd, 1.0f);

    // 5) jf_aug = jf @ augW^T + augb  -> f32
    gemm_nt_bf16<false, true><<<dim3(Dd / BN, Bq / BM), 256, 0, stream>>>(
        jf_b, augW_b, jf_aug, augb, Bq, Dd, Dd, 1.0f);

    // 6) S = (Q @ K^T) / sqrt(D)  (8192 x 2048, K=1024) -> f32
    gemm_nt_bf16<false, false><<<dim3(Nc / BN, Bq / BM), 256, 0, stream>>>(
        Q_b, K_b, S, nullptr, Bq, Nc, Dd, rsqrtD);

    // 7) attn = softmax(S) -> bf16
    softmax_rows<<<Bq / 8, 256, 0, stream>>>(S, attn, Nc);

    // 8) G = attn @ Vt^T = attn @ (Cd*prior)   (8192 x 1024, K=2048) -> f32
    gemm_nt_bf16<false, false><<<dim3(Dd / BN, Bq / BM), 256, 0, stream>>>(
        attn, Vt_b, G, nullptr, Bq, Dd, Nc, 1.0f);

    // 9) combine + LayerNorm -> out
    combine_layernorm<<<Bq / 8, 256, 0, stream>>>(G, jf_aug, balance, gamma, beta, out, Dd);
}
